// QuantMlpWithCudaKernel_78056735638210
// MI455X (gfx1250) — 8-bit matrix operand order repaired and checked by reading, compiled, not run on hardware
//
#include <hip/hip_runtime.h>
#include <stdint.h>


typedef __attribute__((ext_vector_type(2))) int v2i;
typedef __attribute__((ext_vector_type(4))) int v4i;
typedef __attribute__((ext_vector_type(8))) int v8i;

typedef __attribute__((address_space(3))) int8_t lds_i8;

#define QMAX 127.0f
#define QEPS 1e-8f

#define BLK_M   128
#define BLK_N   128
#define KSTEP   64
#define KSTAGE  128
#define BSTRIDE 144

#if __has_builtin(__builtin_amdgcn_sched_barrier)
#define SCHED_FENCE() __builtin_amdgcn_sched_barrier(0)
#else
#define SCHED_FENCE()
#endif

__device__ __forceinline__ void async_g2l_b128(uint32_t lds_off,
                                               const int8_t* gaddr) {
    asm volatile("global_load_async_to_lds_b128 %0, %1, off"
                 :: "v"(lds_off), "v"(gaddr)
                 : "memory");
}
__device__ __forceinline__ void wait_asynccnt0() {
    asm volatile("s_wait_asynccnt 0x0" ::: "memory");
}

__global__ void quant_rows_kernel(const float* __restrict__ in,
                                  int8_t* __restrict__ q,
                                  float* __restrict__ scale,
                                  int cols) {
    const int row = blockIdx.x;
    const float4* r4 = (const float4*)(in + (size_t)row * cols);
    char4* q4 = (char4*)(q + (size_t)row * cols);
    const int cols4 = cols >> 2;

    float m = 0.0f;
    for (int c = threadIdx.x; c < cols4; c += blockDim.x) {
        float4 v = r4[c];
        m = fmaxf(m, fmaxf(fmaxf(fabsf(v.x), fabsf(v.y)),
                           fmaxf(fabsf(v.z), fabsf(v.w))));
    }

    __shared__ float red[256];
    red[threadIdx.x] = m;
    __syncthreads();
    for (int s = 128; s > 0; s >>= 1) {
        if ((int)threadIdx.x < s)
            red[threadIdx.x] = fmaxf(red[threadIdx.x], red[threadIdx.x + s]);
        __syncthreads();
    }
    const float sc = fmaxf(red[0] / QMAX, QEPS);
    if (threadIdx.x == 0) scale[row] = sc;
    const float inv = 1.0f / sc;

    for (int c = threadIdx.x; c < cols4; c += blockDim.x) {
        float4 v = r4[c];
        char4 o;
        o.x = (int8_t)fminf(fmaxf(rintf(v.x * inv), -QMAX), QMAX);
        o.y = (int8_t)fminf(fmaxf(rintf(v.y * inv), -QMAX), QMAX);
        o.z = (int8_t)fminf(fmaxf(rintf(v.z * inv), -QMAX), QMAX);
        o.w = (int8_t)fminf(fmaxf(rintf(v.w * inv), -QMAX), QMAX);
        q4[c] = o;
    }
}

template <bool FUSE_GELU>
__global__ __launch_bounds__(256) void gemm_w8a8_kernel(
        const int8_t* __restrict__ Aq,
        const int8_t* __restrict__ Wq,
        const float* __restrict__ sA,
        const float* __restrict__ sW,
        const float* __restrict__ bias,
        float* __restrict__ Out,
        int M, int Nch, int K) {
    __shared__ __align__(16) int8_t bbuf[2][BLK_N * BSTRIDE];

    const int lane = threadIdx.x & 31;
    const int wave = threadIdx.x >> 5;
    const int lh   = lane & 15;
    const int hi   = lane >> 4;

    const int tilesN = Nch / BLK_N;
    const int bm = blockIdx.x / tilesN;
    const int bn = blockIdx.x % tilesN;
    const int tileM = bm * BLK_M + wave * 16;
    const int tileN = bn * BLK_N;

    auto stageB = [&](int k, int buf) {
        int chunk = wave * 32 + lane;
#pragma unroll
        for (int i = 0; i < 4; ++i, chunk += 256) {
            const int c  = chunk >> 3;
            const int ko = (chunk & 7) * 16;
            const int8_t* src = Wq + (size_t)(tileN + c) * K + k + ko;
            lds_i8* lp = (lds_i8*)&bbuf[buf][c * BSTRIDE + ko];
            async_g2l_b128((uint32_t)(uintptr_t)lp, src);
        }
    };

    const int8_t* aptr = Aq + (size_t)(tileM + lh) * K + (hi << 4);
    auto loadA = [&](int k) -> v8i {
        v4i a0 = *(const v4i*)(aptr + k +  0);
        v4i a1 = *(const v4i*)(aptr + k + 32);
        v8i a = { a0[0], a0[1], a0[2], a0[3], a1[0], a1[1], a1[2], a1[3] };
        return a;
    };

    v8i acc[8] = {};

    auto computeStage = [&](int buf, v8i alo, v8i ahi) {
        SCHED_FENCE();
        const int8_t* base = &bbuf[buf][lh * BSTRIDE + (hi << 4)];
#pragma unroll
        for (int f = 0; f < 16; ++f) {
            const int8_t* bb = base + (f & 7) * (16 * BSTRIDE) + (f >> 3) * 64;
            v4i b0 = *(const v4i*)(bb);
            v4i b1 = *(const v4i*)(bb + 32);
            v8i b = (v8i){ b0[0], b0[1], b0[2], b0[3],
                           b1[0], b1[1], b1[2], b1[3] };
            const v8i a = (f < 8) ? alo : ahi;
            acc[f & 7] = __builtin_amdgcn_wmma_i32_16x16x64_iu8(
                 true, a,  true, b, acc[f & 7],
                 false,  false);
        }
#if __has_builtin(__builtin_amdgcn_sched_group_barrier)
        __builtin_amdgcn_sched_group_barrier(0x100, 4, 0);
#pragma unroll
        for (int i = 0; i < 14; ++i) {
            __builtin_amdgcn_sched_group_barrier(0x008, 1, 0);
            __builtin_amdgcn_sched_group_barrier(0x100, 2, 0);
        }
        __builtin_amdgcn_sched_group_barrier(0x008, 2, 0);
#endif
        SCHED_FENCE();
    };

    stageB(0, 0);
    v8i a0l = loadA(0), a0h = loadA(KSTEP);
    v8i a1l, a1h;
    wait_asynccnt0();
    __syncthreads();

    for (int k = 0; k < K; k += 2 * KSTAGE) {
        stageB(k + KSTAGE, 1);
        a1l = loadA(k + KSTAGE);
        a1h = loadA(k + KSTAGE + KSTEP);
        computeStage(0, a0l, a0h);
        wait_asynccnt0();
        __syncthreads();

        if (k + 2 * KSTAGE < K) {
            stageB(k + 2 * KSTAGE, 0);
            a0l = loadA(k + 2 * KSTAGE);
            a0h = loadA(k + 2 * KSTAGE + KSTEP);
        }
        computeStage(1, a1l, a1h);
        wait_asynccnt0();
        __syncthreads();
    }

    const int rbase = tileM + (hi << 3);
    float sa[8];
#pragma unroll
    for (int v = 0; v < 8; ++v) sa[v] = sA[rbase + v];

#pragma unroll
    for (int t = 0; t < 8; ++t) {
        const int col = tileN + t * 16 + lh;
        const float swc = sW[col];
        const float bc  = bias[col];
#pragma unroll
        for (int v = 0; v < 8; ++v) {
            float val = (float)acc[t][v] * sa[v] * swc + bc;
            if (FUSE_GELU) {
                val = 0.5f * val * (1.0f + erff(val * 0.70710678118654752f));
            }
            Out[(size_t)(rbase + v) * Nch + col] = val;
        }
    }
}

extern "C" void kernel_launch(void* const* d_in, const int* in_sizes, int n_in,
                              void* d_out, int out_size, void* d_ws, size_t ws_size,
                              hipStream_t stream) {
    (void)in_sizes; (void)n_in; (void)out_size; (void)ws_size;

    constexpr int N = 4096, D = 2048, H = 8192;

    const float* x  = (const float*)d_in[0];
    const float* w1 = (const float*)d_in[1];
    const float* b1 = (const float*)d_in[2];
    const float* w2 = (const float*)d_in[3];
    const float* b2 = (const float*)d_in[4];
    float* out = (float*)d_out;

    char* p = (char*)d_ws;
    auto alloc = [&](size_t bytes) {
        char* r = p;
        p += (bytes + 255) & ~(size_t)255;
        return r;
    };
    int8_t* w1q = (int8_t*)alloc((size_t)H * D);
    int8_t* w2q = (int8_t*)alloc((size_t)D * H);
    int8_t* xq  = (int8_t*)alloc((size_t)N * D);
    int8_t* hq  = (int8_t*)alloc((size_t)N * H);
    float*  hbuf = (float*)alloc((size_t)N * H * 4);
    float*  sw1 = (float*)alloc((size_t)H * 4);
    float*  sw2 = (float*)alloc((size_t)D * 4);
    float*  sx  = (float*)alloc((size_t)N * 4);
    float*  sh  = (float*)alloc((size_t)N * 4);

    quant_rows_kernel<<<H, 256, 0, stream>>>(w1, w1q, sw1, D);
    quant_rows_kernel<<<D, 256, 0, stream>>>(w2, w2q, sw2, H);
    quant_rows_kernel<<<N, 256, 0, stream>>>(x,  xq,  sx,  D);

    gemm_w8a8_kernel<true><<<dim3((N / BLK_M) * (H / BLK_N)), 256, 0, stream>>>(
        xq, w1q, sx, sw1, b1, hbuf, N, H, D);

    quant_rows_kernel<<<N, 256, 0, stream>>>(hbuf, hq, sh, H);

    gemm_w8a8_kernel<false><<<dim3((N / BLK_M) * (D / BLK_N)), 256, 0, stream>>>(
        hq, w2q, sh, sw2, b2, out, N, D, H);
}
